// MCASSR_66400194396543
// MI455X (gfx1250) — compile-verified, matrix operand order checked by reading, not run on hardware
//
#include <hip/hip_runtime.h>

// ---------------- problem constants ----------------
#define CCH   64
#define HH    64
#define WWD   64
#define QQ    4096
#define CKN   576          // C*9
#define KINR  580          // CK + 4
#define KPAD  640          // layer-1 K padded to multiple of 64
#define NPAD2 640          // layer-2 N padded to multiple of 128
#define HID   4096
#define MROWS 8192         // B*Q
#define NSHIFT 4
#define EPSV  1e-6f

typedef __attribute__((ext_vector_type(16))) __bf16 v16bf;
typedef __attribute__((ext_vector_type(8)))  __bf16 v8bf;
typedef __attribute__((ext_vector_type(8)))  float  v8f;

__device__ __forceinline__ float clampf(float v, float lo, float hi) {
    return fminf(fmaxf(v, lo), hi);
}
__device__ __forceinline__ int nearest_idx(float c, int n) {
    int i = (int)floorf(((c + 1.0f) * (float)n - 1.0f) * 0.5f + 0.5f);
    return i < 0 ? 0 : (i > n - 1 ? n - 1 : i);
}

// ---------------- weight prep (all weights stored TRANSPOSED [N][K], zero-padded) ----------------
__global__ void pack_w1T(const float* __restrict__ kW1, const float* __restrict__ vW1,
                         __bf16* __restrict__ WT) {
    int idx = blockIdx.x * 256 + threadIdx.x;
    if (idx >= 2 * HID * KPAD) return;
    int n = idx / KPAD;
    int k = idx - n * KPAD;
    float v = 0.f;
    if (k < KINR) v = (n < HID) ? kW1[(size_t)k * HID + n] : vW1[(size_t)k * HID + (n - HID)];
    WT[idx] = (__bf16)v;
}

// W2 [HID][CKN] -> W2T [NPAD2][HID], zero rows past CKN
__global__ void pack_w2T(const float* __restrict__ W2, __bf16* __restrict__ WT) {
    int idx = blockIdx.x * 256 + threadIdx.x;
    if (idx >= NPAD2 * HID) return;
    int n = idx >> 12;               // / HID
    int k = idx & (HID - 1);
    WT[idx] = (__bf16)((n < CKN) ? W2[(size_t)k * CKN + n] : 0.f);
}

// qW1 [CKN][HID] -> qW1T [HID][CKN]
__global__ void pack_qW1T(const float* __restrict__ W, __bf16* __restrict__ WT) {
    int idx = blockIdx.x * 256 + threadIdx.x;
    if (idx >= HID * CKN) return;
    int n = idx / CKN;
    int k = idx - n * CKN;
    WT[idx] = (__bf16)W[(size_t)k * HID + n];
}

__global__ void pack_bias_cat(const float* __restrict__ kb1, const float* __restrict__ vb1,
                              float* __restrict__ bc) {
    int i = blockIdx.x * 256 + threadIdx.x;
    if (i < 2 * HID) bc[i] = (i < HID) ? kb1[i] : vb1[i - HID];
}

__global__ void pad_bias(const float* __restrict__ b, float* __restrict__ bp, int n, int npad) {
    int i = blockIdx.x * 256 + threadIdx.x;
    if (i < npad) bp[i] = (i < n) ? b[i] : 0.f;
}

// ---------------- gather: key/value input rows (one wave per row) ----------------
__global__ void gather_kv(const float* __restrict__ feature, const float* __restrict__ coord,
                          const float* __restrict__ scale,
                          __bf16* __restrict__ Xkv, __bf16* __restrict__ kvbf) {
    int row  = blockIdx.x * 8 + (threadIdx.x >> 5);     // 0 .. 4*8192-1
    int lane = threadIdx.x & 31;
    int s  = row >> 13;
    int bq = row & (MROWS - 1);
    int b  = bq >> 12;
    float vx = (s < 2) ? -1.f : 1.f;
    float vy = (s & 1) ? 1.f : -1.f;

    float cy = coord[(size_t)bq * 2 + 0];
    float cx = coord[(size_t)bq * 2 + 1];
    float sy = scale[(size_t)bq * 2 + 0];
    float sx = scale[(size_t)bq * 2 + 1];
    float tx = ((float)(HH - 1)) / (1.0f - scale[(size_t)b * QQ * 2 + 0]);
    float ty = ((float)(WWD - 1)) / (1.0f - scale[(size_t)b * QQ * 2 + 1]);

    float c0 = clampf(cy + vx / tx + EPSV, -1.f + 1e-6f, 1.f - 1e-6f);
    float c1 = clampf(cx + vy / ty + EPSV, -1.f + 1e-6f, 1.f - 1e-6f);
    int iy = nearest_idx(c0, HH);
    int ix = nearest_idx(c1, WWD);
    float qcy = -1.f + (2.f * iy + 1.f) / (float)HH;
    float qcx = -1.f + (2.f * ix + 1.f) / (float)WWD;
    float rel0 = (cy - qcy) * (float)HH;
    float rel1 = (cx - qcx) * (float)WWD;

    const float* fb = feature + (size_t)b * CCH * HH * WWD;
    __bf16* xrow = Xkv  + (size_t)row * KPAD;
    __bf16* krow = kvbf + (size_t)row * NPAD2;
    for (int cc = lane; cc < CKN; cc += 32) {
        int c = cc / 9, p = cc - c * 9;
        int di = p / 3 - 1, dj = p - (p / 3) * 3 - 1;
        int y = iy + di, x = ix + dj;
        float v = (y >= 0 && y < HH && x >= 0 && x < WWD) ? fb[((size_t)c * HH + y) * WWD + x] : 0.f;
        __bf16 bv = (__bf16)v;
        xrow[cc] = bv;
        krow[cc] = bv;
    }
    if (lane == 0) {
        xrow[CKN + 0] = (__bf16)rel0;
        xrow[CKN + 1] = (__bf16)rel1;
        xrow[CKN + 2] = (__bf16)(sy * (float)HH);
        xrow[CKN + 3] = (__bf16)(sx * (float)WWD);
    }
    for (int cc = KINR + lane; cc < KPAD; cc += 32) xrow[cc] = (__bf16)0.f;
}

// ---------------- gather: local-ensemble query (one wave per row) ----------------
__global__ void gather_query(const float* __restrict__ feature, const float* __restrict__ coord,
                             float* __restrict__ query) {
    int row  = blockIdx.x * 8 + (threadIdx.x >> 5);     // 0 .. 8191
    int lane = threadIdx.x & 31;
    int b = row >> 12;
    float cy = coord[(size_t)row * 2 + 0];
    float cx = coord[(size_t)row * 2 + 1];
    const float rx = 1.f / (float)HH, ry = 1.f / (float)WWD;

    int iys[4], ixs[4];
    float areas[4];
    #pragma unroll
    for (int e = 0; e < 4; e++) {
        float vx = (e < 2) ? -1.f : 1.f;
        float vy = (e & 1) ? 1.f : -1.f;
        float c0 = clampf(cy + vx * rx + EPSV, -1.f + 1e-6f, 1.f - 1e-6f);
        float c1 = clampf(cx + vy * ry + EPSV, -1.f + 1e-6f, 1.f - 1e-6f);
        int iy = nearest_idx(c0, HH), ix = nearest_idx(c1, WWD);
        iys[e] = iy; ixs[e] = ix;
        float qcy = -1.f + (2.f * iy + 1.f) / (float)HH;
        float qcx = -1.f + (2.f * ix + 1.f) / (float)WWD;
        float r0 = (cy - qcy) * (float)HH, r1 = (cx - qcx) * (float)WWD;
        areas[e] = fabsf(r0 * r1) + 1e-9f;
    }
    float tot = areas[0] + areas[1] + areas[2] + areas[3];
    float wts[4];
    #pragma unroll
    for (int e = 0; e < 4; e++) wts[e] = areas[3 - e] / tot;   // diagonal swap

    const float* fb = feature + (size_t)b * CCH * HH * WWD;
    float* qrow = query + (size_t)row * CKN;
    for (int cc = lane; cc < CKN; cc += 32) {
        int c = cc / 9, p = cc - c * 9;
        int di = p / 3 - 1, dj = p - (p / 3) * 3 - 1;
        float acc = 0.f;
        #pragma unroll
        for (int e = 0; e < 4; e++) {
            int y = iys[e] + di, x = ixs[e] + dj;
            float v = (y >= 0 && y < HH && x >= 0 && x < WWD) ? fb[((size_t)c * HH + y) * WWD + x] : 0.f;
            acc += wts[e] * v;
        }
        qrow[cc] = acc;
    }
}

// ---------------- WMMA GEMM: C[M,Npad] = act(A[M,K] @ BT[Npad,K]^T + bias) (*emul) ----------------
// Both A and BT are K-contiguous. M, N multiples of 128; K multiple of 64. No guards.
// block = 256 threads = 8 waves (4 in M x 2 in N); tile 128x128, K-step 64,
// double-buffered LDS staged via GLOBAL_LOAD_ASYNC_TO_LDS_B128 (ASYNCcnt).
template<bool RELU, bool BF16OUT, bool EMUL>
__global__ __launch_bounds__(256) void gemm_bf16_wmma(
    const __bf16* __restrict__ A, int lda,
    const __bf16* __restrict__ BT, int ldbt,
    void* __restrict__ Cout, int ldc,
    const float* __restrict__ bias,
    const __bf16* __restrict__ emul, int lde,
    int K) {
    constexpr int TK = 64, LDP = TK + 8;
    __shared__ __bf16 As[2][128 * LDP];
    __shared__ __bf16 Bs[2][128 * LDP];

    const int tid   = threadIdx.x;
    const int lane  = tid & 31;
    const int waveM = (tid >> 5) & 3;
    const int waveN = tid >> 7;
    const int blockM = blockIdx.y * 128;
    const int blockN = blockIdx.x * 128;
    const int rl   = lane & 15;
    const int hsel = (lane >> 4) << 3;        // K-half select per 16-bit A/B lane layout

    const __bf16* Ab = A  + (size_t)blockM * lda;
    const __bf16* Bb = BT + (size_t)blockN * ldbt;

    // DMA a 128x64 bf16 tile straight into LDS (per-lane 16B), bypassing VGPRs.
    auto load_tile_async = [&](const __bf16* __restrict__ G, int ld, __bf16* dst, int k0) {
        #pragma unroll
        for (int i = 0; i < 4; i++) {
            int idx = tid + i * 256;          // 0..1023
            int r   = idx >> 3;               // 0..127
            int kc  = (idx & 7) << 3;         // 0..56
            unsigned int       ldsa = (unsigned int)(size_t)(dst + r * LDP + kc);
            unsigned long long ga   = (unsigned long long)(size_t)(G + (size_t)r * ld + k0 + kc);
            asm volatile("global_load_async_to_lds_b128 %0, %1, off"
                         :: "v"(ldsa), "v"(ga) : "memory");
        }
    };

    v8f acc[2][4] = {};

    load_tile_async(Ab, lda,  As[0], 0);
    load_tile_async(Bb, ldbt, Bs[0], 0);
    asm volatile("s_wait_asynccnt 0x0" ::: "memory");
    __syncthreads();

    int buf = 0;
    for (int k0 = 0; k0 < K; k0 += TK) {
        int cur = buf;
        buf ^= 1;
        if (k0 + TK < K) {
            if (k0 + 2 * TK < K) {   // prefetch tile after next -> global_prefetch_b8
                __builtin_prefetch(Ab + (size_t)(tid >> 1) * lda + (k0 + 2 * TK), 0, 1);
                __builtin_prefetch(Bb + (size_t)(tid >> 1) * ldbt + (k0 + 2 * TK), 0, 1);
            }
            load_tile_async(Ab, lda,  As[buf], k0 + TK);
            load_tile_async(Bb, ldbt, Bs[buf], k0 + TK);
        }
        #pragma unroll
        for (int kh = 0; kh < TK; kh += 32) {
            v16bf afrag[2], bfrag[4];
            #pragma unroll
            for (int mf = 0; mf < 2; mf++) {
                const __bf16* p = &As[cur][(waveM * 32 + mf * 16 + rl) * LDP + kh + hsel];
                v8bf lo = *(const v8bf*)p;
                v8bf hi = *(const v8bf*)(p + 16);
                #pragma unroll
                for (int j = 0; j < 8; j++) { afrag[mf][j] = lo[j]; afrag[mf][j + 8] = hi[j]; }
            }
            #pragma unroll
            for (int nf = 0; nf < 4; nf++) {
                const __bf16* p = &Bs[cur][(waveN * 64 + nf * 16 + rl) * LDP + kh + hsel];
                v8bf lo = *(const v8bf*)p;
                v8bf hi = *(const v8bf*)(p + 16);
                #pragma unroll
                for (int j = 0; j < 8; j++) { bfrag[nf][j] = lo[j]; bfrag[nf][j + 8] = hi[j]; }
            }
            #pragma unroll
            for (int mf = 0; mf < 2; mf++)
                #pragma unroll
                for (int nf = 0; nf < 4; nf++)
                    acc[mf][nf] = __builtin_amdgcn_wmma_f32_16x16x32_bf16(
                        false, afrag[mf], false, bfrag[nf], (short)0, acc[mf][nf], false, false);
        }
        asm volatile("s_wait_asynccnt 0x0" ::: "memory");
        __syncthreads();
    }

    // epilogue: VGPR r -> row base+r (lanes 0-15) / base+r+8 (lanes 16-31); col = base+(lane&15)
    const int rowAdd = (lane >> 4) << 3;
    #pragma unroll
    for (int mf = 0; mf < 2; mf++) {
        #pragma unroll
        for (int nf = 0; nf < 4; nf++) {
            int col = blockN + waveN * 64 + nf * 16 + rl;
            float bv = bias[col];
            #pragma unroll
            for (int r = 0; r < 8; r++) {
                int row = blockM + waveM * 32 + mf * 16 + rowAdd + r;
                float v = acc[mf][nf][r] + bv;
                if (RELU) v = v > 0.f ? v : 0.f;
                if (EMUL) v *= (float)emul[(size_t)row * lde + col];
                if (BF16OUT) ((__bf16*)Cout)[(size_t)row * ldc + col] = (__bf16)v;
                else         ((float*)Cout)[(size_t)row * ldc + col] = v;
            }
        }
    }
}

// ---------------- attention combine (one wave per row) ----------------
__global__ void attn_combine(const float* __restrict__ query, const float* __restrict__ pk,
                             const float* __restrict__ pv, __bf16* __restrict__ xq) {
    int row  = blockIdx.x * 8 + (threadIdx.x >> 5);
    int lane = threadIdx.x & 31;
    float p0 = 0, p1 = 0, p2 = 0, p3 = 0;
    const float* qr = query + (size_t)row * CKN;
    for (int cc = lane; cc < CKN; cc += 32) {
        float qv = qr[cc];
        p0 += qv * pk[((size_t)0 * MROWS + row) * NPAD2 + cc];
        p1 += qv * pk[((size_t)1 * MROWS + row) * NPAD2 + cc];
        p2 += qv * pk[((size_t)2 * MROWS + row) * NPAD2 + cc];
        p3 += qv * pk[((size_t)3 * MROWS + row) * NPAD2 + cc];
    }
    #pragma unroll
    for (int off = 16; off > 0; off >>= 1) {
        p0 += __shfl_xor(p0, off, 32);
        p1 += __shfl_xor(p1, off, 32);
        p2 += __shfl_xor(p2, off, 32);
        p3 += __shfl_xor(p3, off, 32);
    }
    float mx = fmaxf(fmaxf(p0, p1), fmaxf(p2, p3));
    float e0 = __expf(p0 - mx), e1 = __expf(p1 - mx), e2 = __expf(p2 - mx), e3 = __expf(p3 - mx);
    float inv = 1.f / (e0 + e1 + e2 + e3);
    float w0 = e0 * inv, w1 = e1 * inv, w2 = e2 * inv, w3 = e3 * inv;
    for (int cc = lane; cc < CKN; cc += 32) {
        float acc = w0 * pv[((size_t)0 * MROWS + row) * NPAD2 + cc]
                  + w1 * pv[((size_t)1 * MROWS + row) * NPAD2 + cc]
                  + w2 * pv[((size_t)2 * MROWS + row) * NPAD2 + cc]
                  + w3 * pv[((size_t)3 * MROWS + row) * NPAD2 + cc];
        xq[(size_t)row * CKN + cc] = (__bf16)acc;
    }
}

// ---------------- final tiny layer: [8192,4096] @ [4096,3] + b ----------------
__global__ void final_mlp2(const __bf16* __restrict__ h, const float* __restrict__ W2,
                           const float* __restrict__ b2, float* __restrict__ out) {
    int row  = blockIdx.x * 8 + (threadIdx.x >> 5);
    int lane = threadIdx.x & 31;
    float a0 = 0, a1 = 0, a2 = 0;
    const __bf16* hr = h + (size_t)row * HID;
    for (int k = lane; k < HID; k += 32) {
        float hv = (float)hr[k];
        a0 += hv * W2[k * 3 + 0];
        a1 += hv * W2[k * 3 + 1];
        a2 += hv * W2[k * 3 + 2];
    }
    #pragma unroll
    for (int off = 16; off > 0; off >>= 1) {
        a0 += __shfl_xor(a0, off, 32);
        a1 += __shfl_xor(a1, off, 32);
        a2 += __shfl_xor(a2, off, 32);
    }
    if (lane == 0) {
        out[(size_t)row * 3 + 0] = a0 + b2[0];
        out[(size_t)row * 3 + 1] = a1 + b2[1];
        out[(size_t)row * 3 + 2] = a2 + b2[2];
    }
}

// ---------------- launcher ----------------
extern "C" void kernel_launch(void* const* d_in, const int* in_sizes, int n_in,
                              void* d_out, int out_size, void* d_ws, size_t ws_size,
                              hipStream_t stream) {
    (void)in_sizes; (void)n_in; (void)out_size; (void)ws_size;
    const float* feature = (const float*)d_in[0];
    const float* coord   = (const float*)d_in[1];
    const float* scale   = (const float*)d_in[2];
    const float* kW1 = (const float*)d_in[3];
    const float* kb1 = (const float*)d_in[4];
    const float* kW2 = (const float*)d_in[5];
    const float* kb2 = (const float*)d_in[6];
    const float* vW1 = (const float*)d_in[7];
    const float* vb1 = (const float*)d_in[8];
    const float* vW2 = (const float*)d_in[9];
    const float* vb2 = (const float*)d_in[10];
    const float* qW1 = (const float*)d_in[11];
    const float* qb1 = (const float*)d_in[12];
    const float* qW2 = (const float*)d_in[13];
    const float* qb2 = (const float*)d_in[14];
    float* out = (float*)d_out;

    char* ws = (char*)d_ws;
    size_t off = 0;
    auto take = [&](size_t bytes) -> char* {
        char* p = ws + off;
        off = (off + bytes + 255) & ~(size_t)255;
        return p;
    };
    __bf16* Xkv    = (__bf16*)take((size_t)NSHIFT * MROWS * KPAD * 2);
    __bf16* kvbf   = (__bf16*)take((size_t)NSHIFT * MROWS * NPAD2 * 2);
    float*  query  = (float*) take((size_t)MROWS * CKN * 4);
    __bf16* Wc1T   = (__bf16*)take((size_t)2 * HID * KPAD * 2);
    float*  bc1    = (float*) take((size_t)2 * HID * 4);
    __bf16* kW2T   = (__bf16*)take((size_t)NPAD2 * HID * 2);
    __bf16* vW2T   = (__bf16*)take((size_t)NPAD2 * HID * 2);
    __bf16* qW1T   = (__bf16*)take((size_t)HID * CKN * 2);
    float*  kb2p   = (float*) take((size_t)NPAD2 * 4);
    float*  vb2p   = (float*) take((size_t)NPAD2 * 4);
    float*  pk     = (float*) take((size_t)NSHIFT * MROWS * NPAD2 * 4);
    float*  pv     = (float*) take((size_t)NSHIFT * MROWS * NPAD2 * 4);
    __bf16* hidden = (__bf16*)take((size_t)MROWS * 2 * HID * 2);
    __bf16* xq     = (__bf16*)take((size_t)MROWS * CKN * 2);

    // weight prep (transposed + padded)
    pack_w1T<<<(2 * HID * KPAD + 255) / 256, 256, 0, stream>>>(kW1, vW1, Wc1T);
    pack_bias_cat<<<(2 * HID + 255) / 256, 256, 0, stream>>>(kb1, vb1, bc1);
    pack_w2T<<<(NPAD2 * HID + 255) / 256, 256, 0, stream>>>(kW2, kW2T);
    pack_w2T<<<(NPAD2 * HID + 255) / 256, 256, 0, stream>>>(vW2, vW2T);
    pack_qW1T<<<(HID * CKN + 255) / 256, 256, 0, stream>>>(qW1, qW1T);
    pad_bias<<<(NPAD2 + 255) / 256, 256, 0, stream>>>(kb2, kb2p, CKN, NPAD2);
    pad_bias<<<(NPAD2 + 255) / 256, 256, 0, stream>>>(vb2, vb2p, CKN, NPAD2);

    // gathers
    gather_kv<<<NSHIFT * MROWS / 8, 256, 0, stream>>>(feature, coord, scale, Xkv, kvbf);
    gather_query<<<MROWS / 8, 256, 0, stream>>>(feature, coord, query);

    // per-shift MLPs through WMMA GEMMs
    for (int s = 0; s < NSHIFT; s++) {
        const __bf16* Arow = Xkv + (size_t)s * MROWS * KPAD;
        dim3 g1(2 * HID / 128, MROWS / 128);
        gemm_bf16_wmma<true, true, false><<<g1, 256, 0, stream>>>(
            Arow, KPAD, Wc1T, KPAD, hidden, 2 * HID, bc1, nullptr, 0, KPAD);
        dim3 g2(NPAD2 / 128, MROWS / 128);
        gemm_bf16_wmma<false, false, true><<<g2, 256, 0, stream>>>(
            hidden, 2 * HID, kW2T, HID, pk + (size_t)s * MROWS * NPAD2, NPAD2, kb2p,
            kvbf + (size_t)s * MROWS * NPAD2, NPAD2, HID);
        gemm_bf16_wmma<false, false, true><<<g2, 256, 0, stream>>>(
            hidden + HID, 2 * HID, vW2T, HID, pv + (size_t)s * MROWS * NPAD2, NPAD2, vb2p,
            kvbf + (size_t)s * MROWS * NPAD2, NPAD2, HID);
    }

    // attention + final MLP
    attn_combine<<<MROWS / 8, 256, 0, stream>>>(query, pk, pv, xq);

    dim3 g3(HID / 128, MROWS / 128);
    gemm_bf16_wmma<true, true, false><<<g3, 256, 0, stream>>>(
        xq, CKN, qW1T, CKN, hidden, HID, qb1, nullptr, 0, CKN);

    final_mlp2<<<MROWS / 8, 256, 0, stream>>>(hidden, qW2, qb2, out);
}
